// BG_GRU_18726057410810
// MI455X (gfx1250) — compile-verified, matrix operand order checked by reading, not run on hardware
//
#include <hip/hip_runtime.h>
#include <hip/hip_bf16.h>

#define GRU_H 512
#define GRU_B 512
#define GRU_T 128

// LDS B-tile: 48 rows (3 gates x 16 cols), padded stride to avoid bank conflicts
#define BS_STRIDE 520   // halves; 1040 B per row, lane l -> banks 4l..4l+3

typedef __attribute__((ext_vector_type(16))) _Float16 v16h;
typedef __attribute__((ext_vector_type(8)))  float    v8f;

struct alignas(16) Half8 { _Float16 h[8]; };

// Load one 16x16x32 f16 WMMA fragment row (works for global A and LDS B).
// ISA layout: lane<16 -> K = k0+{0..7, 16..23}; lane>=16 -> +8.
// 'kb' already includes the per-lane +0/+8. Two aligned 16B loads per lane.
__device__ __forceinline__ v16h load_frag(const _Float16* __restrict__ rowp, int kb) {
    Half8 lo = *(const Half8*)(rowp + kb);
    Half8 hi = *(const Half8*)(rowp + kb + 16);
    v16h f;
#pragma unroll
    for (int i = 0; i < 8; ++i) { f[i] = lo.h[i]; f[i + 8] = hi.h[i]; }
    return f;
}

// One-time prep: w_hh f32 -> f16 (row-major, 3H x H), and zero-init h buffers.
__global__ void gru_prep(const float* __restrict__ w_hh,
                         _Float16* __restrict__ w16,
                         float* __restrict__ h32,
                         _Float16* __restrict__ h16) {
    int idx = blockIdx.x * blockDim.x + threadIdx.x;
    if (idx < 3 * GRU_H * GRU_H) w16[idx] = (_Float16)w_hh[idx];
    if (idx < GRU_B * GRU_H) { h32[idx] = 0.0f; h16[idx] = (_Float16)0.0f; }
}

// Fused GRU timestep.
// Block = one n-tile (16 hidden cols) x 256 batch rows; 8 waves, each a 32Mx16N tile.
// B (w_hh rows for cols {j, j+H, j+2H}, full K) staged in LDS, shared by all 8 waves.
// Per wave: 6 f32 accumulators (2 M-subtiles x 3 gates), A double-buffered from global.
__global__ __launch_bounds__(256) void gru_step(
    const _Float16* __restrict__ h16_in,
    const float*    __restrict__ h32_in,
    _Float16*       __restrict__ h16_out,
    float*          __restrict__ h32_out,
    const _Float16* __restrict__ w16,     // (3H, H) row-major f16
    const float*    __restrict__ x,       // (B, T)
    const float*    __restrict__ wi,      // (3H)
    const float*    __restrict__ b_ih,    // (3H)
    const float*    __restrict__ b_hh,    // (3H)
    int t)
{
    __shared__ alignas(16) _Float16 Bs[48 * BS_STRIDE];   // 49,920 B

    const int tid  = threadIdx.x;
    const int lane = tid & 31;
    const int wave = tid >> 5;
    const int nt   = blockIdx.x & 31;          // n-tile 0..31
    const int mblk = blockIdx.x >> 5;          // 0..1
    const int n0   = nt * 16;
    const int wbase = mblk * 256 + wave * 32;  // this wave's 32-row M range

    // ---- Cooperative stage of B tile into LDS: 48 rows x 512 halves ----
    // LDS row r (gate = r>>4, col = n0 + (r&15)) <- w16 row (gate*H + n0 + (r&15))
#pragma unroll
    for (int i = tid; i < 48 * 64; i += 256) {             // 64 x 16B chunks per row
        const int row = i >> 6;
        const int c   = i & 63;
        const int grow = (row >> 4) * GRU_H + n0 + (row & 15);
        const uint4 v = *(const uint4*)(w16 + (size_t)grow * GRU_H + c * 8);
        *(uint4*)(&Bs[row * BS_STRIDE + c * 8]) = v;
    }
    __syncthreads();

    const int lrow  = lane & 15;
    const int khalf = (lane >> 4) << 3;        // +8 for upper half-wave

    const _Float16* arow0 = h16_in + (size_t)(wbase + lrow) * GRU_H;
    const _Float16* arow1 = arow0 + (size_t)16 * GRU_H;
    const _Float16* bls   = &Bs[lrow * BS_STRIDE];

    v8f acc[6] = {};                           // [sub*3 + gate]

    // ---- K loop: fully unrolled, A fragments double-buffered ----
    v16h ac0 = load_frag(arow0, khalf);
    v16h ac1 = load_frag(arow1, khalf);
#pragma unroll
    for (int kk = 0; kk < 16; ++kk) {
        const int kb = kk * 32 + khalf;
        v16h an0 = ac0, an1 = ac1;
        if (kk < 15) {                          // prefetch next K-slice of A
            an0 = load_frag(arow0, kb + 32);
            an1 = load_frag(arow1, kb + 32);
        }
#pragma unroll
        for (int g = 0; g < 3; ++g) {
            const v16h bf = load_frag(bls + g * (16 * BS_STRIDE), kb);
            acc[g]     = __builtin_amdgcn_wmma_f32_16x16x32_f16(
                             false, ac0, false, bf, (short)0, acc[g],     false, false);
            acc[3 + g] = __builtin_amdgcn_wmma_f32_16x16x32_f16(
                             false, ac1, false, bf, (short)0, acc[3 + g], false, false);
        }
        ac0 = an0; ac1 = an1;
    }

    // ---- Fused gates + h update ----
    // C/D layout: column = n0 + (lane&15); vgpr i -> row subtile_base + (lane>=16?8:0) + i
    const int col = n0 + lrow;
    const float wir = wi[col], wiz = wi[GRU_H + col], win = wi[2 * GRU_H + col];
    const float bir = b_ih[col], biz = b_ih[GRU_H + col], bin_ = b_ih[2 * GRU_H + col];
    const float bhr = b_hh[col], bhz = b_hh[GRU_H + col], bhn  = b_hh[2 * GRU_H + col];

#pragma unroll
    for (int sub = 0; sub < 2; ++sub) {
        const int rbase = wbase + sub * 16 + khalf;
#pragma unroll
        for (int i = 0; i < 8; ++i) {
            const int brow = rbase + i;
            const float xt = x[brow * GRU_T + t];
            const float rg = 1.0f / (1.0f + __expf(-(xt * wir + bir + acc[sub * 3 + 0][i] + bhr)));
            const float zg = 1.0f / (1.0f + __expf(-(xt * wiz + biz + acc[sub * 3 + 1][i] + bhz)));
            const float ng = tanhf(xt * win + bin_ + rg * (acc[sub * 3 + 2][i] + bhn));
            const float hv = (1.0f - zg) * ng + zg * h32_in[(size_t)brow * GRU_H + col];
            h32_out[(size_t)brow * GRU_H + col] = hv;
            h16_out[(size_t)brow * GRU_H + col] = (_Float16)hv;
        }
    }
}

// out[b] = relu(h[b,:]) . fc_w + fc_b ; one wave (32 lanes) per row.
__global__ __launch_bounds__(256) void gru_fc(const float* __restrict__ h32,
                                              const float* __restrict__ fc_w,
                                              const float* __restrict__ fc_b,
                                              float* __restrict__ out) {
    const int lane = threadIdx.x & 31;
    const int wave = threadIdx.x >> 5;
    const int row  = blockIdx.x * 8 + wave;     // 0..511
    float s = 0.0f;
#pragma unroll
    for (int j = lane; j < GRU_H; j += 32) {
        const float hv = h32[(size_t)row * GRU_H + j];
        s += fmaxf(hv, 0.0f) * fc_w[j];
    }
#pragma unroll
    for (int off = 16; off > 0; off >>= 1) s += __shfl_xor(s, off, 32);
    if (lane == 0) out[row] = s + fc_b[0];
}

extern "C" void kernel_launch(void* const* d_in, const int* in_sizes, int n_in,
                              void* d_out, int out_size, void* d_ws, size_t ws_size,
                              hipStream_t stream) {
    const float* x    = (const float*)d_in[0];   // (512,128)
    const float* w_ih = (const float*)d_in[1];   // (1536,1) -> wi
    const float* w_hh = (const float*)d_in[2];   // (1536,512)
    const float* b_ih = (const float*)d_in[3];   // (1536)
    const float* b_hh = (const float*)d_in[4];   // (1536)
    const float* fc_w = (const float*)d_in[5];   // (512)
    const float* fc_b = (const float*)d_in[6];   // (1)
    float* out = (float*)d_out;

    // Workspace layout (4.5 MB total):
    char* ws = (char*)d_ws;
    _Float16* w16  = (_Float16*)(ws);                                   // 1,572,864 B
    float*    h32a = (float*)   (ws + 1572864);                         // 1 MB
    float*    h32b = (float*)   (ws + 1572864 + 1048576);               // 1 MB
    _Float16* h16a = (_Float16*)(ws + 1572864 + 2 * 1048576);           // 512 KB
    _Float16* h16b = (_Float16*)(ws + 1572864 + 2 * 1048576 + 524288);  // 512 KB

    gru_prep<<<3072, 256, 0, stream>>>(w_hh, w16, h32a, h16a);

    for (int t = 0; t < GRU_T; ++t) {
        const bool even = (t & 1) == 0;
        gru_step<<<64, 256, 0, stream>>>(
            even ? h16a : h16b, even ? h32a : h32b,
            even ? h16b : h16a, even ? h32b : h32a,
            w16, x, w_ih, b_ih, b_hh, t);
    }
    // t=127 (odd) writes buffers 'a' -> final h in h32a
    gru_fc<<<64, 256, 0, stream>>>(h32a, fc_w, fc_b, out);
}
